// CausalSelfAttention_274877906956
// MI455X (gfx1250) — matrix operand order repaired and checked by reading, not compiled, not run on hardware
//
#include <hip/hip_runtime.h>
#include <math.h>

#define DD 1024
#define SS 2048
#define BB 4

typedef unsigned short u16;
typedef __attribute__((ext_vector_type(8)))  u16      v8u;
typedef __attribute__((ext_vector_type(16))) u16      v16u;
typedef __attribute__((ext_vector_type(16))) _Float16 v16h;
typedef __attribute__((ext_vector_type(8)))  float    v8f;
typedef __attribute__((ext_vector_type(4)))  unsigned v4ui;
typedef __attribute__((ext_vector_type(8)))  unsigned v8ui;

__device__ __forceinline__ u16 f2h(float f) {
    union { _Float16 h; u16 u; } c; c.h = (_Float16)f; return c.u;
}
__device__ __forceinline__ v16h as_h(v16u v) {
    union { v16u u; v16h h; } c; c.u = v; return c.h;
}
__device__ __forceinline__ v8f wmma_f16(v16u a, v16u b, v8f c) {
    return __builtin_amdgcn_wmma_f32_16x16x32_f16(false, as_h(a), false, as_h(b),
                                                  (short)0, c, false, false);
}
// Assemble an A-fragment (16-bit 16x32 layout): 16 contiguous K per lane-half (k = 16h + s).
__device__ __forceinline__ v16u a_frag(const u16* p) {
    v8u a0 = *(const v8u*)p;
    v8u a1 = *(const v8u*)(p + 8);
    v16u av;
    #pragma unroll
    for (int i = 0; i < 8; ++i) { av[i] = a0[i]; av[i + 8] = a1[i]; }
    return av;
}
__device__ __forceinline__ float rmax16(float v) {
    v = fmaxf(v, __shfl_xor(v, 8, 16));
    v = fmaxf(v, __shfl_xor(v, 4, 16));
    v = fmaxf(v, __shfl_xor(v, 2, 16));
    v = fmaxf(v, __shfl_xor(v, 1, 16));
    return v;
}
__device__ __forceinline__ float rsum16(float v) {
    v += __shfl_xor(v, 8, 16);
    v += __shfl_xor(v, 4, 16);
    v += __shfl_xor(v, 2, 16);
    v += __shfl_xor(v, 1, 16);
    return v;
}

// ---------------------------------------------------------------- convert
__global__ void cvt_kernel(const float* __restrict__ x,  const float* __restrict__ wq,
                           const float* __restrict__ wk, const float* __restrict__ wv,
                           u16* xb, u16* wqb, u16* wkb, u16* wvb) {
    const long NX = (long)BB * SS * DD;
    const long NW = (long)DD * DD;
    long i = (long)blockIdx.x * blockDim.x + threadIdx.x;
    if (i < NX)            xb [i]          = f2h(x [i]);
    else if (i < NX+NW)    wqb[i-NX]       = f2h(wq[i-NX]);
    else if (i < NX+2*NW)  wkb[i-NX-NW]    = f2h(wk[i-NX-NW]);
    else if (i < NX+3*NW)  wvb[i-NX-2*NW]  = f2h(wv[i-NX-2*NW]);
}

// ---------------------------------------------------------------- QKV GEMM
// y = x @ W^T.  One wave per 64x64 output tile (4x4 register blocking,
// 16 v8f accumulators). 512 WMMAs per wave; ~32 FLOP/B from L2.
// Q,K stored row-major [b][s][e]; V stored transposed Vt[b][e][s].
__global__ __launch_bounds__(256) void qkv_kernel(
        const u16* __restrict__ xb,
        const u16* __restrict__ wqb, const u16* __restrict__ wkb, const u16* __restrict__ wvb,
        u16* __restrict__ Qb, u16* __restrict__ Kb, u16* __restrict__ Vt) {
    int lane = threadIdx.x & 31;
    int hi   = (lane >> 4) & 1;
    int n    = lane & 15;
    int wg   = blockIdx.x * 8 + (threadIdx.x >> 5);   // global wave id, 0..6143
    int et   = wg & 15;   wg >>= 4;                   // e-tile (64 wide) 0..15
    int st   = wg & 31;   wg >>= 5;                   // s-tile (64 tall) 0..31
    int b    = wg & 3;    wg >>= 2;
    int mat  = wg;                                    // 0=Q 1=K 2=V
    const u16* W = (mat == 0) ? wqb : (mat == 1) ? wkb : wvb;
    int sbase = st * 64, ebase = et * 64;

    const u16* arow = xb + ((long)b * SS + sbase + n) * DD + hi * 16;
    const u16* brow = W + (long)(ebase + n) * DD + hi * 16;

    v8f acc[16];
    #pragma unroll
    for (int t = 0; t < 16; ++t) acc[t] = (v8f){0,0,0,0,0,0,0,0};

    for (int c = 0; c < DD / 32; ++c) {
        int d0 = c * 32;
        v16u a[4], bf[4];
        #pragma unroll
        for (int i = 0; i < 4; ++i) {
            a[i]  = a_frag(arow + (long)i * 16 * DD + d0);
            bf[i] = *(const v16u*)(brow + (long)i * 16 * DD + d0);
        }
        #pragma unroll
        for (int mi = 0; mi < 4; ++mi)
            #pragma unroll
            for (int ni = 0; ni < 4; ++ni)
                acc[mi * 4 + ni] = wmma_f16(a[mi], bf[ni], acc[mi * 4 + ni]);
    }

    if (mat < 2) {
        u16* out = ((mat == 0) ? Qb : Kb) + (long)b * SS * DD;
        #pragma unroll
        for (int mi = 0; mi < 4; ++mi)
            #pragma unroll
            for (int ni = 0; ni < 4; ++ni)
                #pragma unroll
                for (int r = 0; r < 8; ++r) {
                    int M = mi * 16 + r + hi * 8;
                    out[(long)(sbase + M) * DD + ebase + ni * 16 + n] = f2h(acc[mi*4+ni][r]);
                }
    } else {
        u16* out = Vt + (long)b * DD * SS;
        #pragma unroll
        for (int mi = 0; mi < 4; ++mi)
            #pragma unroll
            for (int ni = 0; ni < 4; ++ni)
                #pragma unroll
                for (int r = 0; r < 8; ++r) {
                    int M = mi * 16 + r + hi * 8;
                    out[(long)(ebase + ni * 16 + n) * SS + sbase + M] = f2h(acc[mi*4+ni][r]);
                }
    }
}

// ---------------------------------------------------------------- flash attention
// One WG (8 waves) per (batch, 16-query-row block). Q tile staged into LDS by the
// Tensor Data Mover (wave 0 issues tensor_load_to_lds, waits TENSORcnt). Per
// 128-wide k-block: wave w computes S cols [w*16,w*16+16), cross-wave online
// softmax via LDS, then wave w accumulates its O slice d in [w*128,(w+1)*128).
__global__ __launch_bounds__(256) void attn_kernel(
        const u16* __restrict__ Qb, const u16* __restrict__ Kb,
        const u16* __restrict__ Vt, float* __restrict__ out) {
    __shared__ __align__(16) u16  sQ[16 * DD];    // 32 KB query tile
    __shared__ __align__(16) u16  sP[16 * 128];   // 4 KB probability tile (f16)
    __shared__ float sMax[8][16];
    __shared__ float sSum[8][16];

    int tid  = threadIdx.x;
    int lane = tid & 31, w = tid >> 5;
    int hi   = (lane >> 4) & 1, n = lane & 15;
    int bq   = blockIdx.x;
    int b    = bq >> 7;             // 128 q-tiles per batch
    int qt   = bq & 127;
    int qbase = qt * 16;

    // ---- stage Q tile (16 x 1024 f16, row stride DD) into LDS via TDM
    if (w == 0) {
        unsigned      lds_off = (unsigned)(size_t)(&sQ[0]);
        unsigned long ga      = (unsigned long)(size_t)(Qb + ((long)b * SS + qbase) * DD);
        v4ui g0;
        g0[0] = 0x1u;                                   // count=1 (valid user D#)
        g0[1] = lds_off;                                // lds_addr (bytes)
        g0[2] = (unsigned)(ga & 0xFFFFFFFFu);           // global_addr[31:0]
        g0[3] = (unsigned)((ga >> 32) & 0x01FFFFFFu)    // global_addr[56:32]
              | (2u << 30);                             // type=2 (image)
        v8ui g1;
        g1[0] = (1u << 16);            // workgroup_mask=0, data_size=1 (2 bytes)
        g1[1] = ((unsigned)DD) << 16;  // tensor_dim0 = 1024 (bits 79:48)
        g1[2] = (16u << 16);           // tensor_dim1 = 16   (bits 111:80)
        g1[3] = ((unsigned)DD) << 16;  // tile_dim0   = 1024 (bits 127:112)
        g1[4] = 16u;                   // tile_dim1   = 16, tile_dim2 = 0
        g1[5] = (unsigned)DD;          // tensor_dim0_stride = 1024 (bits 207:160)
        g1[6] = 0u;
        g1[7] = 0u;                    // tensor_dim1_stride = 0 (2D tile)
        asm volatile("tensor_load_to_lds %0, %1" :: "s"(g0), "s"(g1) : "memory");
        __builtin_amdgcn_s_wait_tensorcnt(0);
    }
    __syncthreads();

    float m[8], l[8];
    #pragma unroll
    for (int r = 0; r < 8; ++r) { m[r] = -INFINITY; l[r] = 0.f; }
    v8f o[8];
    #pragma unroll
    for (int j = 0; j < 8; ++j) o[j] = (v8f){0,0,0,0,0,0,0,0};

    const float scale = 0.03125f;   // 1/sqrt(1024)
    const u16* Kbb = Kb + (long)b * SS * DD;
    const u16* Vtb = Vt + (long)b * DD * SS;
    int nkb = (qbase + 15) / 128 + 1;

    for (int kb = 0; kb < nkb; ++kb) {
        int kbase = kb * 128;

        // prefetch next k-block K rows toward the WGP while we compute
        if (kb + 1 < nkb)
            __builtin_prefetch(Kbb + (long)(kbase + 128 + w * 16 + n) * DD, 0, 0);

        // ---- S tile: Q (LDS) x K^T (global), contract D=1024
        v8f s = (v8f){0,0,0,0,0,0,0,0};
        const u16* arow = sQ + n * DD + hi * 16;
        const u16* krow = Kbb + (long)(kbase + w * 16 + n) * DD + hi * 16;
        #pragma unroll 8
        for (int c = 0; c < DD / 32; ++c) {
            int d0 = c * 32;
            v16u av = a_frag(arow + d0);
            v16u bv = *(const v16u*)(krow + d0);
            s = wmma_f16(av, bv, s);
        }

        // ---- causal mask + scale, per-wave row max
        int kcol = kbase + w * 16 + n;
        #pragma unroll
        for (int r = 0; r < 8; ++r) {
            int qg = qbase + r + hi * 8;
            float v = s[r] * scale;
            v = (kcol <= qg) ? v : -INFINITY;
            s[r] = v;
            float t = rmax16(v);
            if (n == 0) sMax[w][r + hi * 8] = t;
        }
        __syncthreads();

        // ---- new running max + rescale factor (per row, replicated per lane-half)
        float nm[8], f[8];
        #pragma unroll
        for (int r = 0; r < 8; ++r) {
            int row = r + hi * 8;
            float t = m[r];
            #pragma unroll
            for (int w2 = 0; w2 < 8; ++w2) t = fmaxf(t, sMax[w2][row]);
            nm[r] = t;
            f[r]  = __expf(m[r] - t);   // m=-inf first pass -> 0
        }

        // ---- P = exp(s - nm): write f16 tile to LDS, per-wave row sums
        #pragma unroll
        for (int r = 0; r < 8; ++r) {
            float p = __expf(s[r] - nm[r]);  // fully-masked -> exp(-inf)=0
            sP[(r + hi * 8) * 128 + w * 16 + n] = f2h(p);
            float t = rsum16(p);
            if (n == 0) sSum[w][r + hi * 8] = t;
        }
        __syncthreads();

        // ---- update l, m; rescale O
        #pragma unroll
        for (int r = 0; r < 8; ++r) {
            int row = r + hi * 8;
            float t = 0.f;
            #pragma unroll
            for (int w2 = 0; w2 < 8; ++w2) t += sSum[w2][row];
            l[r] = l[r] * f[r] + t;
            m[r] = nm[r];
        }
        #pragma unroll
        for (int j = 0; j < 8; ++j)
            #pragma unroll
            for (int r = 0; r < 8; ++r) o[j][r] *= f[r];

        // ---- O += P (LDS) x V (global Vt), contract k=128
        int dw = w * 128;
        #pragma unroll
        for (int j = 0; j < 8; ++j) {
            v8f acc = o[j];
            const u16* prow = sP + n * 128 + hi * 16;
            const u16* vrow = Vtb + (long)(dw + j * 16 + n) * SS + kbase + hi * 16;
            #pragma unroll
            for (int c = 0; c < 4; ++c) {
                v16u av = a_frag(prow + c * 32);
                v16u bv = *(const v16u*)(vrow + c * 32);
                acc = wmma_f16(av, bv, acc);
            }
            o[j] = acc;
        }
        __syncthreads();   // protect sP/sMax/sSum before next iteration
    }

    // ---- epilogue: normalize and store fp32
    float* ob = out + ((long)b * SS + qbase) * DD;
    #pragma unroll
    for (int r = 0; r < 8; ++r) {
        float inv = 1.0f / l[r];
        int M = r + hi * 8;
        #pragma unroll
        for (int j = 0; j < 8; ++j)
            ob[(long)M * DD + w * 128 + j * 16 + n] = o[j][r] * inv;
    }
}

// ---------------------------------------------------------------- launch
extern "C" void kernel_launch(void* const* d_in, const int* in_sizes, int n_in,
                              void* d_out, int out_size, void* d_ws, size_t ws_size,
                              hipStream_t stream) {
    const float* x  = (const float*)d_in[0];
    const float* wq = (const float*)d_in[1];
    const float* wk = (const float*)d_in[2];
    const float* wv = (const float*)d_in[3];

    char* ws = (char*)d_ws;
    const long NX = (long)BB * SS * DD;      // 8,388,608
    const long NW = (long)DD * DD;           // 1,048,576
    u16* xb  = (u16*)(ws);
    u16* wqb = (u16*)(ws + 2 * NX);
    u16* wkb = (u16*)(ws + 2 * (NX + NW));
    u16* wvb = (u16*)(ws + 2 * (NX + 2 * NW));
    u16* Qb  = (u16*)(ws + 2 * (NX + 3 * NW));
    u16* Kb  = Qb + NX;
    u16* Vt  = Kb + NX;

    long total = NX + 3 * NW;
    cvt_kernel<<<(int)((total + 255) / 256), 256, 0, stream>>>(x, wq, wk, wv,
                                                               xb, wqb, wkb, wvb);
    // 3 mats * 4 batches * 32 * 16 (64x64) tiles = 6144 waves / 8 per WG
    qkv_kernel<<<768, 256, 0, stream>>>(xb, wqb, wkb, wvb, Qb, Kb, Vt);
    // B * S/16 = 512 workgroups
    attn_kernel<<<512, 256, 0, stream>>>(Qb, Kb, Vt, (float*)d_out);
}
